// MPNNBlock_42726334660740
// MI455X (gfx1250) — compile-verified, matrix operand order checked by reading, not run on hardware
//
#include <hip/hip_runtime.h>
#include <hip/hip_bf16.h>

#define N_NODES  100000
#define N_EDGES  1600000
#define HIDDEN   128
#define STEPS    4
#define N_GRAPHS 64
#define KP       (HIDDEN / 2)      // 64 packed k-pairs per row
#define OUT3H    (3 * HIDDEN)      // 384
#define LDS_LD   68                // LDS row stride in dwords (68 % 64 == 4 -> conflict-free)

typedef __attribute__((ext_vector_type(16))) __bf16 v16bf;
typedef __attribute__((ext_vector_type(8)))  float  v8f;

__device__ __forceinline__ unsigned short f2bf(float f) {
    unsigned int u = __float_as_uint(f);
    unsigned int r = (u + 0x7FFFu + ((u >> 16) & 1u)) >> 16;   // RNE
    return (unsigned short)r;
}

__device__ __forceinline__ unsigned int pack2bf(float lo, float hi) {
    return (unsigned int)f2bf(lo) | ((unsigned int)f2bf(hi) << 16);
}

__device__ __forceinline__ v8f wmma_bf16(v16bf a, v16bf b, v8f c) {
    return __builtin_amdgcn_wmma_f32_16x16x32_bf16(
        false, a, false, b, (short)0, c, false, false);
}

// Stage a 16-row x 64-dword (16x128 bf16) A tile into LDS, coalesced.
// 256 threads, each moves one uint4: row = tid>>4, col = (tid&15)*4.
__device__ __forceinline__ void stage_a(const unsigned int* __restrict__ g, int row0,
                                        unsigned int* sA, int tid) {
    int row = tid >> 4;
    int col = (tid & 15) * 4;
    uint4 v = *(const uint4*)(g + (size_t)(row0 + row) * KP + col);
    *(uint4*)(sA + row * LDS_LD + col) = v;
}

// A operand from LDS tile: operand dword i (0..7), lane-half `half`:
//   dword = kk*16 + (i<4 ? 0 : 8) + (i&3) + half*4  -> two ds_load_b128.
__device__ __forceinline__ v16bf load_a_lds(const unsigned int* sA, int kk, int lane) {
    int m = lane & 15, half = lane >> 4;
    const unsigned int* p = sA + m * LDS_LD + kk * 16 + half * 4;
    union { v16bf v; uint4 q[2]; } r;
    r.q[0] = *(const uint4*)(p);        // K pairs 0..3 of this half
    r.q[1] = *(const uint4*)(p + 8);    // K pairs 8..11 of this half
    return r.v;
}

// B operand: k-pair-major packed weights Bp[kp][n] (ldn dwords per kp row).
__device__ __forceinline__ v16bf load_b_pk(const unsigned int* __restrict__ Bp,
                                           int ldn, int kk, int col0, int lane) {
    int n = lane & 15, half = lane >> 4;
    union { v16bf v; unsigned int u[8]; } r;
#pragma unroll
    for (int i = 0; i < 8; ++i) {
        int kp = kk * 16 + ((i & 4) ? 8 : 0) + (i & 3) + half * 4;
        r.u[i] = Bp[(size_t)kp * ldn + col0 + n];
    }
    return r.v;
}

// ---------------- one-time prep: pack weights into B layout ------------------
__global__ void prep_w_kernel(const float* __restrict__ W, unsigned int* __restrict__ Wp) {
    int idx = blockIdx.x * blockDim.x + threadIdx.x;
    if (idx >= STEPS * KP * HIDDEN) return;
    int s  = idx / (KP * HIDDEN);
    int kp = (idx / HIDDEN) % KP;
    int n  = idx % HIDDEN;
    const float* Ws = W + (size_t)s * HIDDEN * HIDDEN;
    Wp[idx] = pack2bf(Ws[(size_t)(2 * kp) * HIDDEN + n],
                      Ws[(size_t)(2 * kp + 1) * HIDDEN + n]);
}

__global__ void prep_wt_kernel(const float* __restrict__ W, unsigned int* __restrict__ Bp) {
    int idx = blockIdx.x * blockDim.x + threadIdx.x;
    if (idx >= KP * OUT3H) return;
    int kp = idx / OUT3H;
    int j  = idx % OUT3H;
    const float* p = W + (size_t)j * HIDDEN + 2 * kp;
    Bp[idx] = pack2bf(p[0], p[1]);
}

// ---------------- f32 -> packed bf16 shadow copy -----------------------------
__global__ void cvt_kernel(const float* __restrict__ src, unsigned int* __restrict__ dst,
                           int n_pairs) {
    int i = blockIdx.x * blockDim.x + threadIdx.x;
    if (i >= n_pairs) return;
    float2 v = ((const float2*)src)[i];
    dst[i] = pack2bf(v.x, v.y);
}

// ---------------- m = h @ W[s] ----------------------------------------------
__global__ void gemm_m_kernel(const unsigned int* __restrict__ h_bf,
                              const unsigned int* __restrict__ Wp,
                              float* __restrict__ m_out) {
    __shared__ unsigned int sA[16 * LDS_LD];
    int tid  = threadIdx.x;
    int lane = tid & 31;
    int wave = tid >> 5;               // 0..7 -> 16-col tile
    int row0 = blockIdx.x * 16;
    int col0 = wave * 16;
    __builtin_prefetch(Wp + col0, 0, 3);

    stage_a(h_bf, row0, sA, tid);
    __syncthreads();

    v8f acc = {};
#pragma unroll
    for (int kk = 0; kk < 4; ++kk) {
        v16bf a = load_a_lds(sA, kk, lane);
        v16bf b = load_b_pk(Wp, HIDDEN, kk, col0, lane);
        acc = wmma_bf16(a, b, acc);
    }
    int n = lane & 15, half = lane >> 4;
#pragma unroll
    for (int i = 0; i < 8; ++i)
        m_out[(size_t)(row0 + i + 8 * half) * HIDDEN + col0 + n] = acc[i];
}

// ------------- aggr[dst] += m[src], one wave per edge ------------------------
__global__ void scatter_kernel(const float* __restrict__ m,
                               const long long* __restrict__ src,
                               const long long* __restrict__ dst,
                               float* __restrict__ aggr) {
    int e = blockIdx.x * 8 + (threadIdx.x >> 5);
    if (e >= N_EDGES) return;
    int lane = threadIdx.x & 31;
    long long s = src[e], d = dst[e];
    const float4 v = *(const float4*)(m + (size_t)s * HIDDEN + lane * 4);
    float* out = aggr + (size_t)d * HIDDEN + lane * 4;
    atomicAdd(out + 0, v.x);
    atomicAdd(out + 1, v.y);
    atomicAdd(out + 2, v.z);
    atomicAdd(out + 3, v.w);
}

// ------------------------------- GRU cell ------------------------------------
// One block = one 16-node tile; wave w owns columns [16w,16w+16) of each gate.
// A tiles (aggr, h) staged once in LDS and shared by all 8 waves.
__global__ void gru_kernel(const unsigned int* __restrict__ aggr_bf,
                           const unsigned int* __restrict__ h_bf,
                           float* __restrict__ h,
                           const unsigned int* __restrict__ Bih,
                           const unsigned int* __restrict__ Bhh,
                           const float* __restrict__ b_ih,
                           const float* __restrict__ b_hh) {
    __shared__ unsigned int sAm[16 * LDS_LD];
    __shared__ unsigned int sAh[16 * LDS_LD];
    int tid  = threadIdx.x;
    int lane = tid & 31;
    int wave = tid >> 5;               // 0..7
    int row0 = blockIdx.x * 16;
    int j0   = wave * 16;
    int n = lane & 15, half = lane >> 4;
    __builtin_prefetch(Bih + j0, 0, 3);
    __builtin_prefetch(Bhh + j0, 0, 3);

    stage_a(aggr_bf, row0, sAm, tid);
    stage_a(h_bf,    row0, sAh, tid);
    __syncthreads();

    v8f ir = {}, iz = {}, inn = {}, hr = {}, hz = {}, hn = {};
#pragma unroll
    for (int kk = 0; kk < 4; ++kk) {
        v16bf am = load_a_lds(sAm, kk, lane);
        v16bf ah = load_a_lds(sAh, kk, lane);
        ir  = wmma_bf16(am, load_b_pk(Bih, OUT3H, kk, 0 * HIDDEN + j0, lane), ir);
        iz  = wmma_bf16(am, load_b_pk(Bih, OUT3H, kk, 1 * HIDDEN + j0, lane), iz);
        inn = wmma_bf16(am, load_b_pk(Bih, OUT3H, kk, 2 * HIDDEN + j0, lane), inn);
        hr  = wmma_bf16(ah, load_b_pk(Bhh, OUT3H, kk, 0 * HIDDEN + j0, lane), hr);
        hz  = wmma_bf16(ah, load_b_pk(Bhh, OUT3H, kk, 1 * HIDDEN + j0, lane), hz);
        hn  = wmma_bf16(ah, load_b_pk(Bhh, OUT3H, kk, 2 * HIDDEN + j0, lane), hn);
    }

    int c = j0 + n;
    float bir = b_ih[c],              bhr = b_hh[c];
    float biz = b_ih[HIDDEN + c],     bhz = b_hh[HIDDEN + c];
    float bin = b_ih[2 * HIDDEN + c], bhn = b_hh[2 * HIDDEN + c];

#pragma unroll
    for (int i = 0; i < 8; ++i) {
        size_t off = (size_t)(row0 + i + 8 * half) * HIDDEN + c;
        float hv = h[off];
        float r  = 1.0f / (1.0f + expf(-(ir[i] + bir + hr[i] + bhr)));
        float z  = 1.0f / (1.0f + expf(-(iz[i] + biz + hz[i] + bhz)));
        float nn = tanhf(inn[i] + bin + r * (hn[i] + bhn));
        h[off] = (1.0f - z) * nn + z * hv;
    }
}

// ---------------- ReLU + per-graph mean pool ---------------------------------
__global__ void pool_kernel(const float* __restrict__ h,
                            const long long* __restrict__ batch,
                            float* __restrict__ sums,
                            float* __restrict__ counts) {
    size_t idx = (size_t)blockIdx.x * blockDim.x + threadIdx.x;
    if (idx >= (size_t)N_NODES * HIDDEN) return;
    int node = (int)(idx >> 7);
    int c    = (int)(idx & (HIDDEN - 1));
    long long g = batch[node];
    float v = h[idx];
    v = v > 0.0f ? v : 0.0f;
    atomicAdd(&sums[(size_t)g * HIDDEN + c], v);
    if (c == 0) atomicAdd(&counts[g], 1.0f);
}

__global__ void finalize_kernel(const float* __restrict__ sums,
                                const float* __restrict__ counts,
                                float* __restrict__ out) {
    int idx = blockIdx.x * blockDim.x + threadIdx.x;
    if (idx >= N_GRAPHS * HIDDEN) return;
    float cnt = counts[idx / HIDDEN];
    out[idx] = sums[idx] / fmaxf(cnt, 1.0f);
}

extern "C" void kernel_launch(void* const* d_in, const int* in_sizes, int n_in,
                              void* d_out, int out_size, void* d_ws, size_t ws_size,
                              hipStream_t stream) {
    (void)in_sizes; (void)n_in; (void)out_size; (void)ws_size;

    const float*     x      = (const float*)d_in[0];
    const long long* edge   = (const long long*)d_in[1];   // [2, N_EDGES] int64
    const long long* batch  = (const long long*)d_in[2];   // [N_NODES] int64
    const float*     weight = (const float*)d_in[3];       // [STEPS, H, H]
    const float*     w_ih   = (const float*)d_in[4];       // [3H, H]
    const float*     w_hh   = (const float*)d_in[5];       // [3H, H]
    const float*     b_ih   = (const float*)d_in[6];
    const float*     b_hh   = (const float*)d_in[7];
    float*           out    = (float*)d_out;

    const size_t NH = (size_t)N_NODES * HIDDEN;
    float* ws     = (float*)d_ws;
    float* hbuf   = ws;                 // N x H f32
    float* mbuf   = ws + NH;            // N x H f32
    float* aggr   = ws + 2 * NH;        // N x H f32
    float* sums   = ws + 3 * NH;        // G x H f32
    float* counts = sums + (size_t)N_GRAPHS * HIDDEN;     // G f32

    unsigned int* h_bf    = (unsigned int*)(counts + N_GRAPHS);  // N x KP dwords
    unsigned int* aggr_bf = h_bf + NH / 2;                        // N x KP dwords
    unsigned int* Wp      = aggr_bf + NH / 2;                     // STEPS*KP*H
    unsigned int* Bih     = Wp + (size_t)STEPS * KP * HIDDEN;     // KP*3H
    unsigned int* Bhh     = Bih + (size_t)KP * OUT3H;             // KP*3H

    prep_w_kernel<<<(STEPS * KP * HIDDEN + 255) / 256, 256, 0, stream>>>(weight, Wp);
    prep_wt_kernel<<<(KP * OUT3H + 255) / 256, 256, 0, stream>>>(w_ih, Bih);
    prep_wt_kernel<<<(KP * OUT3H + 255) / 256, 256, 0, stream>>>(w_hh, Bhh);

    hipMemcpyAsync(hbuf, x, NH * sizeof(float), hipMemcpyDeviceToDevice, stream);
    cvt_kernel<<<(int)(NH / 2 + 255) / 256, 256, 0, stream>>>(hbuf, h_bf, (int)(NH / 2));

    const int row_tiles = N_NODES / 16;           // 6250, exact
    for (int s = 0; s < STEPS; ++s) {
        gemm_m_kernel<<<row_tiles, 256, 0, stream>>>(
            h_bf, Wp + (size_t)s * KP * HIDDEN, mbuf);
        hipMemsetAsync(aggr, 0, NH * sizeof(float), stream);
        scatter_kernel<<<(N_EDGES + 7) / 8, 256, 0, stream>>>(
            mbuf, edge, edge + N_EDGES, aggr);
        cvt_kernel<<<(int)(NH / 2 + 255) / 256, 256, 0, stream>>>(aggr, aggr_bf, (int)(NH / 2));
        gru_kernel<<<row_tiles, 256, 0, stream>>>(
            aggr_bf, h_bf, hbuf, Bih, Bhh, b_ih, b_hh);
        cvt_kernel<<<(int)(NH / 2 + 255) / 256, 256, 0, stream>>>(hbuf, h_bf, (int)(NH / 2));
    }

    hipMemsetAsync(sums, 0, (size_t)(N_GRAPHS * HIDDEN + N_GRAPHS) * sizeof(float), stream);
    pool_kernel<<<(int)((NH + 255) / 256), 256, 0, stream>>>(hbuf, batch, sums, counts);
    finalize_kernel<<<(N_GRAPHS * HIDDEN + 255) / 256, 256, 0, stream>>>(sums, counts, out);
}
